// EGT_747324309849
// MI455X (gfx1250) — matrix operand order repaired and checked by reading, not compiled, not run on hardware
//
#include <hip/hip_runtime.h>
#include <hip/hip_bf16.h>

#define DIMN 96
#define NH   32
#define HD   3

typedef __attribute__((ext_vector_type(16))) __bf16 v16bf;
typedef __attribute__((ext_vector_type(8)))  float  v8f;
typedef __attribute__((ext_vector_type(2)))  __bf16 v2bf;

union Frag16 { v16bf v; unsigned int u[8]; };

#if __has_builtin(__builtin_amdgcn_cvt_pk_bf16_f32)
__device__ __forceinline__ unsigned int pack2bf(float lo, float hi) {
  union { v2bf v; unsigned int u; } cv;
  cv.v = __builtin_amdgcn_cvt_pk_bf16_f32(lo, hi);
  return cv.u;
}
#else
__device__ __forceinline__ unsigned int f2bf(float x) {
  unsigned int u = __float_as_uint(x);
  return (u + 0x7FFFu + ((u >> 16) & 1u)) >> 16; // RNE to bf16
}
__device__ __forceinline__ unsigned int pack2bf(float lo, float hi) {
  return f2bf(lo) | (f2bf(hi) << 16);
}
#endif

// ---------------------------------------------------------------------------
// Multi-output GEMM: Co[M,LDC] = act( A[M,K] @ Wo[K,LDC] + bo ) (+ rs*resid)
// for o in 0..NOUT-1. A is staged to LDS ONCE and multiplied against up to 3
// weight tiles (saves A bandwidth for q/k/v and We/Wb). All leading dims are
// compile-time -> immediate-offset addressing. f32 -> bf16 (RNE) -> WMMA.
// Grid: (ceil(M/64), LDC/32). Block: 256 threads = 8 waves (4x2 16x16 tiles).
// Tail rows of A are CLAMPED (row m of A only affects row m of C; OOB rows
// masked at store). Interior blocks take an unguarded store fast path.
// ---------------------------------------------------------------------------
#define BM 64
#define BN 32
#define BK 32
#define LSTR 40   // LDS row stride in halves (80B): 8B-aligned, conflict-light

template<int K, int LDA, int LDW, int LDC, int RELU, int RES, int NOUT>
__global__ __launch_bounds__(256)
void gemm_bf16_wmma(const float* __restrict__ A,
                    const float* __restrict__ W0, const float* __restrict__ b0, float* __restrict__ C0,
                    const float* __restrict__ W1, const float* __restrict__ b1, float* __restrict__ C1,
                    const float* __restrict__ W2, const float* __restrict__ b2, float* __restrict__ C2,
                    const float* __restrict__ resid, const float* __restrict__ wscale, int M)
{
  __shared__ unsigned short As[BM * LSTR];
  __shared__ unsigned short Bs[NOUT * BN * LSTR];

  const int tid  = threadIdx.x;
  const int lane = tid & 31;
  const int wave = tid >> 5;      // 0..7
  const int wr   = wave & 3;      // 16-row tile within block
  const int wc   = wave >> 2;     // 16-col tile within block
  const int rowBase = blockIdx.x * BM;
  const int colBase = blockIdx.y * BN;
  const int hlf = lane >> 4;      // lane half (0/1)
  const int l16 = lane & 15;

  // A staging: 4 threads per row, 8 consecutive k each (two float4 loads)
  const int ar = tid >> 2;                 // 0..63
  const int ak = (tid & 3) * 8;            // 0,8,16,24
  const int arow = min(rowBase + ar, M - 1);
  const float* aptr = A + (long long)arow * LDA + ak;

  // B staging: thread covers (n, kq..kq+3): 4 coalesced b32 loads, 1 b64 store
  const int bn_s = tid & 31;
  const int bk_s = (tid >> 5) * 4;         // 0..28
  const float* Wl[3] = {W0, W1, W2};
  const float* wps[NOUT];
  #pragma unroll
  for (int o = 0; o < NOUT; ++o) wps[o] = Wl[o] + colBase + bn_s + bk_s * LDW;

  v8f acc[NOUT] = {};

  #pragma unroll
  for (int k0 = 0; k0 < K; k0 += BK) {
    // ---- stage A tile (64x32), f32 -> packed bf16, b64 LDS stores
    const float4* ap4 = (const float4*)(aptr + k0);
    float4 x0 = ap4[0];
    float4 x1 = ap4[1];
    *(uint2*)&As[ar * LSTR + ak]     = make_uint2(pack2bf(x0.x, x0.y), pack2bf(x0.z, x0.w));
    *(uint2*)&As[ar * LSTR + ak + 4] = make_uint2(pack2bf(x1.x, x1.y), pack2bf(x1.z, x1.w));

    // ---- stage W tiles (32x32 each) transposed into [n][k]
    #pragma unroll
    for (int o = 0; o < NOUT; ++o) {
      const float* wp = wps[o] + k0 * LDW;
      float w0v = wp[0 * LDW];
      float w1v = wp[1 * LDW];
      float w2v = wp[2 * LDW];
      float w3v = wp[3 * LDW];
      *(uint2*)&Bs[o * BN * LSTR + bn_s * LSTR + bk_s] =
          make_uint2(pack2bf(w0v, w1v), pack2bf(w2v, w3v));
    }
    __syncthreads();

    // A fragment: 16-bit 16x32 layout (lane: M=l16; half selects K groups)
    Frag16 a;
    const int am = (wr * 16 + l16) * LSTR;
    #pragma unroll
    for (int p = 0; p < 8; ++p) {
      int kb = (p < 4) ? (2 * p + 8 * hlf) : (16 + 2 * (p - 4) + 8 * hlf);
      a.u[p] = *(const unsigned int*)&As[am + kb];
    }
    // B fragment per output: 32x16 layout (lane: N=l16; half selects K groups)
    const int bnb = (wc * 16 + l16) * LSTR;
    #pragma unroll
    for (int o = 0; o < NOUT; ++o) {
      Frag16 b;
      #pragma unroll
      for (int p = 0; p < 8; ++p) {
        b.u[p] = *(const unsigned int*)&Bs[o * BN * LSTR + bnb + ((p < 4) ? (2 * p + 8 * hlf) : (16 + 2 * (p - 4) + 8 * hlf))];
      }
      acc[o] = __builtin_amdgcn_wmma_f32_16x16x32_bf16(
          false, a.v, false, b.v, (short)0, acc[o], false, false);
    }
    __syncthreads();
  }

  // Epilogue. f32 C/D layout: VGPR r -> M = r + 8*half, N = l16.
  // Lane's 8 rows are consecutive: row0..row0+7.
  const int colg = colBase + wc * 16 + l16;
  const int row0 = rowBase + wr * 16 + 8 * hlf;
  float rs = 1.0f;
  if (RES) rs = wscale ? (1.0f + wscale[0]) : 1.0f;
  const float* biases[3] = {b0, b1, b2};
  float* Cs[3] = {C0, C1, C2};
  const bool full = (rowBase + BM <= M);

  #pragma unroll
  for (int o = 0; o < NOUT; ++o) {
    const float bsv = biases[o][colg];
    float* cp = Cs[o] + (long long)row0 * LDC + colg;
    const float* rp = RES ? (resid + (long long)row0 * LDC + colg) : nullptr;
    if (full) {
      #pragma unroll
      for (int r = 0; r < 8; ++r) {
        float v = acc[o][r] + bsv;
        if (RELU) v = fmaxf(v, 0.0f);
        if (RES)  v += rs * rp[r * LDC];
        cp[r * LDC] = v;
      }
    } else {
      #pragma unroll
      for (int r = 0; r < 8; ++r) {
        if (row0 + r < M) {
          float v = acc[o][r] + bsv;
          if (RELU) v = fmaxf(v, 0.0f);
          if (RES)  v += rs * rp[r * LDC];
          cp[r * LDC] = v;
        }
      }
    }
  }
}

// ---------------------------------------------------------------------------
// Row LayerNorm: one wave32 per row, D in {32, 96}
// ---------------------------------------------------------------------------
__global__ __launch_bounds__(256)
void ln_rows(const float* __restrict__ x, const float* __restrict__ g,
             const float* __restrict__ b, float* __restrict__ y,
             long long Mrows, int D)
{
  long long row = (long long)blockIdx.x * 8 + (threadIdx.x >> 5);
  if (row >= Mrows) return;
  const int lane = threadIdx.x & 31;
  const float* xr = x + row * (long long)D;
  float vals[3];
  float s = 0.0f;
  int c = 0;
  for (int j = lane; j < D; j += 32) { float v = xr[j]; vals[c++] = v; s += v; }
  #pragma unroll
  for (int o = 16; o; o >>= 1) s += __shfl_xor(s, o);
  const float mu = s / (float)D;
  float ss = 0.0f; c = 0;
  for (int j = lane; j < D; j += 32) { float d = vals[c++] - mu; ss += d * d; }
  #pragma unroll
  for (int o = 16; o; o >>= 1) ss += __shfl_xor(ss, o);
  const float inv = rsqrtf(ss / (float)D + 1e-5f);
  float* yr = y + row * (long long)D;
  c = 0;
  for (int j = lane; j < D; j += 32) yr[j] = (vals[c++] - mu) * inv * g[j] + b[j];
}

__global__ void zero_u32(unsigned int* __restrict__ p, long long n) {
  long long i = (long long)blockIdx.x * blockDim.x + threadIdx.x;
  if (i < n) p[i] = 0u;
}

// Order-preserving float<->uint mapping for atomic max
__device__ __forceinline__ void atomicMaxF(unsigned int* addr, float v) {
  unsigned int u = __float_as_uint(v);
  u = (u & 0x80000000u) ? ~u : (u | 0x80000000u);
  atomicMax(addr, u);
}
__device__ __forceinline__ float unmapF(unsigned int u) {
  return (u & 0x80000000u) ? __uint_as_float(u & 0x7FFFFFFFu)
                           : __uint_as_float(~u);
}

// ---------------------------------------------------------------------------
// Edge pass 1: score = dot(k[src],q[dst])/2 + B; e_out = score + edge;
// segment max into mmax[dst,h]. `score` holds B on input, score on output.
// Lanes of a wave cover the 32 heads of one edge -> contiguous gathers.
// ---------------------------------------------------------------------------
__global__ __launch_bounds__(256)
void edge_score(const float* __restrict__ q, const float* __restrict__ k,
                const float* __restrict__ edge,
                const int* __restrict__ src, const int* __restrict__ dst,
                float* __restrict__ score, float* __restrict__ e_out,
                unsigned int* __restrict__ mmax, long long EH)
{
  long long i = (long long)blockIdx.x * blockDim.x + threadIdx.x;
  if (i >= EH) return;
  const long long e = i >> 5;
  const int h = (int)(i & 31);
  const int s = src[e], d = dst[e];
  const float* kp = k + (long long)s * DIMN + h * HD;
  const float* qp = q + (long long)d * DIMN + h * HD;
  float sc = (kp[0] * qp[0] + kp[1] * qp[1] + kp[2] * qp[2]) * 0.5f + score[i];
  score[i] = sc;
  e_out[i] = sc + edge[i];
  atomicMaxF(&mmax[(long long)d * NH + h], sc);
}

// ---------------------------------------------------------------------------
// Edge pass 2: ex = exp(score - m[dst]); den[dst,h] += ex;
// aggnum[dst,h,0:3] += ex * v[src,h,:]; aggnum[dst,h,3] += ex * ve
// ---------------------------------------------------------------------------
__global__ __launch_bounds__(256)
void edge_exp_agg(const float* __restrict__ score, const float* __restrict__ v,
                  const float* __restrict__ ve,
                  const int* __restrict__ src, const int* __restrict__ dst,
                  const unsigned int* __restrict__ mmax,
                  float* __restrict__ den, float* __restrict__ agg, long long EH)
{
  long long i = (long long)blockIdx.x * blockDim.x + threadIdx.x;
  if (i >= EH) return;
  const long long e = i >> 5;
  const int h = (int)(i & 31);
  const int s = src[e], d = dst[e];
  const float ex = __expf(score[i] - unmapF(mmax[(long long)d * NH + h]));
  atomicAdd(&den[(long long)d * NH + h], ex);
  const float* vp = v + (long long)s * DIMN + h * HD;
  float* ap = agg + (long long)d * (NH * 4) + h * 4;
  atomicAdd(ap + 0, ex * vp[0]);
  atomicAdd(ap + 1, ex * vp[1]);
  atomicAdd(ap + 2, ex * vp[2]);
  atomicAdd(ap + 3, ex * ve[i]);
}

// agg /= den (per head); empty segments -> 0 (matches segment_sum of empty)
__global__ __launch_bounds__(256)
void agg_norm(float* __restrict__ agg, const float* __restrict__ den, long long n)
{
  long long i = (long long)blockIdx.x * blockDim.x + threadIdx.x;
  if (i >= n) return;
  const long long nd = i >> 7;
  const int h = (int)((i >> 2) & 31);
  const float dn = den[nd * NH + h];
  const float a = agg[i];
  agg[i] = (dn > 0.0f) ? a / dn : 0.0f;
}

// ---------------------------------------------------------------------------
extern "C" void kernel_launch(void* const* d_in, const int* in_sizes, int n_in,
                              void* d_out, int out_size, void* d_ws, size_t ws_size,
                              hipStream_t stream)
{
  const float* feat = (const float*)d_in[0];
  const float* edge = (const float*)d_in[1];
  const float* Wq  = (const float*)d_in[2];  const float* bq  = (const float*)d_in[3];
  const float* Wk  = (const float*)d_in[4];  const float* bk  = (const float*)d_in[5];
  const float* Wv  = (const float*)d_in[6];  const float* bv  = (const float*)d_in[7];
  const float* We  = (const float*)d_in[8];  const float* be  = (const float*)d_in[9];
  const float* Wb  = (const float*)d_in[10]; const float* bb  = (const float*)d_in[11];
  const float* Wvv = (const float*)d_in[12]; const float* bvv = (const float*)d_in[13];
  const float* Wf0 = (const float*)d_in[14]; const float* bf0 = (const float*)d_in[15];
  const float* Wf1 = (const float*)d_in[16]; const float* bf1 = (const float*)d_in[17];
  const float* Wm0 = (const float*)d_in[18]; const float* bm0 = (const float*)d_in[19];
  const float* Wm1 = (const float*)d_in[20]; const float* bm1 = (const float*)d_in[21];
  const float* g_n0 = (const float*)d_in[22]; const float* b_n0 = (const float*)d_in[23];
  const float* g_e0 = (const float*)d_in[24]; const float* b_e0 = (const float*)d_in[25];
  const float* g_e1 = (const float*)d_in[26]; const float* b_e1 = (const float*)d_in[27];
  const float* g_m  = (const float*)d_in[28]; const float* b_m  = (const float*)d_in[29];
  const float* w    = (const float*)d_in[30];
  const int* src = (const int*)d_in[31];
  const int* dst = (const int*)d_in[32];

  const long long N = in_sizes[0] / DIMN;
  const long long E = in_sizes[1] / NH;

  float* out_h = (float*)d_out;                 // [N, 96]
  float* out_e = out_h + N * DIMN;              // [E, 32]

  // workspace carving (f32)
  float* p = (float*)d_ws;
  float* h0  = p; p += N * DIMN;                // LN(feat), later LN(h)
  float* qb  = p; p += N * DIMN;                // q, later MLP hidden [N,192] part 1
  float* kb  = p; p += N * DIMN;                // k, later MLP hidden part 2
  float* vb  = p; p += N * DIMN;                // v
  float* ewb = p; p += E * NH;                  // LN(edge), later LN(e)
  float* veb = p; p += E * NH;                  // ve, later edge-FFN hidden part 1
  float* scb = p; p += E * NH;                  // B -> score, later hidden part 2
  unsigned int* mmax = (unsigned int*)p; p += N * NH;
  float* den = p; p += N * NH;
  float* agg = p; p += N * NH * 4;
  float* tnode = qb;                            // [N, 192] (qb+kb contiguous)
  float* tedge = veb;                           // [E, 64]  (veb+scb contiguous)

  auto cdiv = [](long long a, long long b) -> unsigned int {
    return (unsigned int)((a + b - 1) / b);
  };

  // 1) h0 = LN(feat)
  ln_rows<<<cdiv(N, 8), 256, 0, stream>>>(feat, g_n0, b_n0, h0, N, DIMN);

  // 2) q/k/v = h0 @ {Wq,Wk,Wv} + b : single launch, A staged once (NOUT=3)
  dim3 gN96(cdiv(N, BM), DIMN / BN);
  gemm_bf16_wmma<96,96,96,96,0,0,3><<<gN96, 256, 0, stream>>>(
      h0, Wq, bq, qb, Wk, bk, kb, Wv, bv, vb, nullptr, nullptr, (int)N);

  // 3) ew = LN(edge)
  ln_rows<<<cdiv(E, 8), 256, 0, stream>>>(edge, g_e0, b_e0, ewb, E, NH);

  // 4) {ve, B} = ew @ {We,Wb} + b : single launch (NOUT=2)
  dim3 gE32(cdiv(E, BM), NH / BN);
  gemm_bf16_wmma<32,32,32,32,0,0,2><<<gE32, 256, 0, stream>>>(
      ewb, We, be, veb, Wb, bb, scb, nullptr, nullptr, nullptr, nullptr, nullptr, (int)E);

  // 5) zero mmax/den/agg (contiguous region)
  const long long initN = N * NH + N * NH + N * NH * 4;
  zero_u32<<<cdiv(initN, 256), 256, 0, stream>>>(mmax, initN);

  // 6) score + e_out + segment max
  const long long EH = E * NH;
  edge_score<<<cdiv(EH, 256), 256, 0, stream>>>(qb, kb, edge, src, dst, scb, out_e, mmax, EH);

  // 7) exp / den / weighted scatter
  edge_exp_agg<<<cdiv(EH, 256), 256, 0, stream>>>(scb, vb, veb, src, dst, mmax, den, agg, EH);

  // 8) agg normalize
  const long long NA = N * NH * 4;
  agg_norm<<<cdiv(NA, 256), 256, 0, stream>>>(agg, den, NA);

  // 9) h = feat*(1+w) + agg @ Wvv + bvv
  gemm_bf16_wmma<128,128,96,96,0,1,1><<<gN96, 256, 0, stream>>>(
      agg, Wvv, bvv, out_h, nullptr, nullptr, nullptr, nullptr, nullptr, nullptr, feat, w, (int)N);

  // 10) hn = LN(h)
  ln_rows<<<cdiv(N, 8), 256, 0, stream>>>(out_h, g_m, b_m, h0, N, DIMN);

  // 11) t = relu(hn @ Wm0 + bm0)   [N, 192]
  dim3 gN192(cdiv(N, BM), (2 * DIMN) / BN);
  gemm_bf16_wmma<96,96,192,192,1,0,1><<<gN192, 256, 0, stream>>>(
      h0, Wm0, bm0, tnode, nullptr, nullptr, nullptr, nullptr, nullptr, nullptr, nullptr, nullptr, (int)N);

  // 12) h += t @ Wm1 + bm1
  gemm_bf16_wmma<192,192,96,96,0,1,1><<<gN96, 256, 0, stream>>>(
      tnode, Wm1, bm1, out_h, nullptr, nullptr, nullptr, nullptr, nullptr, nullptr, out_h, nullptr, (int)N);

  // 13) en = LN(e)
  ln_rows<<<cdiv(E, 8), 256, 0, stream>>>(out_e, g_e1, b_e1, ewb, E, NH);

  // 14) tf = relu(en @ Wf0 + bf0)   [E, 64]
  dim3 gE64(cdiv(E, BM), (2 * NH) / BN);
  gemm_bf16_wmma<32,32,64,64,1,0,1><<<gE64, 256, 0, stream>>>(
      ewb, Wf0, bf0, tedge, nullptr, nullptr, nullptr, nullptr, nullptr, nullptr, nullptr, nullptr, (int)E);

  // 15) e += tf @ Wf1 + bf1
  gemm_bf16_wmma<64,64,32,32,0,1,1><<<gE32, 256, 0, stream>>>(
      tedge, Wf1, bf1, out_e, nullptr, nullptr, nullptr, nullptr, nullptr, nullptr, out_e, nullptr, (int)E);
}
